// TopicPropagationModel_45775761440862
// MI455X (gfx1250) — matrix operand order repaired and checked by reading, not compiled, not run on hardware
//
#include <hip/hip_runtime.h>
#include <hip/hip_bf16.h>

#define N      3072
#define D_IN   64
#define HID    128
#define NH     4
#define HD     32
#define L      2
#define B      1024
#define ALPHA  0.2f
#define TD     0.1f
#define EPS    1e-05f

typedef __attribute__((ext_vector_type(16))) _Float16 v16h;
typedef __attribute__((ext_vector_type(8)))  float    v8f;

// ---------------------------------------------------------------- tmax
__global__ void init_tmax_kernel(float* tmax) {
    if (threadIdx.x == 0) *tmax = 0.0f;   // time_matrix in [0,1), positive
}

__global__ void reduce_max_kernel(const float* __restrict__ t, int n, float* out) {
    float m = 0.0f;
    for (int i = blockIdx.x * blockDim.x + threadIdx.x; i < n; i += gridDim.x * blockDim.x)
        m = fmaxf(m, t[i]);
    for (int o = 16; o > 0; o >>= 1) m = fmaxf(m, __shfl_xor(m, o, 32));
    __shared__ float sm[8];
    int lane = threadIdx.x & 31, w = threadIdx.x >> 5;
    if (lane == 0) sm[w] = m;
    __syncthreads();
    if (threadIdx.x == 0) {
        float bm = sm[0];
        for (int i = 1; i < (int)(blockDim.x >> 5); ++i) bm = fmaxf(bm, sm[i]);
        atomicMax((int*)out, __float_as_int(bm));   // positive floats order as ints
    }
}

// ---------------------------------------------------------------- struct = emb @ Ws + bs
__global__ void struct_kernel(const float* __restrict__ emb, const float* __restrict__ Ws,
                              const float* __restrict__ bs, float* __restrict__ x) {
    int idx = blockIdx.x * blockDim.x + threadIdx.x;       // N*HID
    if (idx >= N * HID) return;
    int c = idx & (HID - 1);
    int n = idx >> 7;
    float acc = bs[c];
    const float* er = emb + (size_t)n * D_IN;
#pragma unroll 8
    for (int i = 0; i < D_IN; ++i) acc += er[i] * Ws[i * HID + c];
    x[idx] = acc;
}

// ------------------------------------------------ h = x @ W (per head): fp32 h + f16 h-transposed
__global__ void proj_kernel(const float* __restrict__ x, const float* __restrict__ W,
                            float* __restrict__ h, _Float16* __restrict__ hhT) {
    int idx = blockIdx.x * blockDim.x + threadIdx.x;       // NH*N*HD
    if (idx >= NH * N * HD) return;
    int d    = idx & (HD - 1);
    int n    = (idx / HD) % N;
    int head = idx / (HD * N);
    const float* w  = W + (size_t)head * HID * HD + d;
    const float* xr = x + (size_t)n * HID;
    float acc = 0.0f;
#pragma unroll 8
    for (int i = 0; i < HID; ++i) acc += xr[i] * w[i * HD];
    h[idx] = acc;
    hhT[((size_t)head * HD + d) * N + n] = (_Float16)acc;  // K-contiguous for WMMA B loads
}

// ---------------------------------------------------------------- e_src / e_dst
__global__ void edot_kernel(const float* __restrict__ h, const float* __restrict__ a,
                            float* __restrict__ esrc, float* __restrict__ edst) {
    int idx = blockIdx.x * blockDim.x + threadIdx.x;       // NH*N
    if (idx >= NH * N) return;
    int head = idx / N;
    const float* hr = h + (size_t)idx * HD;
    const float* as = a + head * 2 * HD;                    // gat_a[l,head,:,0]
    float s0 = 0.0f, s1 = 0.0f;
#pragma unroll 8
    for (int d = 0; d < HD; ++d) { s0 += hr[d] * as[d]; s1 += hr[d] * as[HD + d]; }
    esrc[idx] = s0;
    edst[idx] = s1;
}

// ---------------------------------------------------------------- fused attention (flash-style, WMMA)
// One wave per (16-row tile, head). Logits built directly in the 16-bit A-operand
// layout: lane L holds row (L&15); K slots = {kb..kb+7, 16+kb..16+kb+7}, kb = (L<16)?0:8.
// Each lane's 16 j-indices are two contiguous runs of 8 -> 32B vector loads.
__global__ __launch_bounds__(128)
void attn_kernel(const float* __restrict__ adj, const float* __restrict__ tmat,
                 const float* __restrict__ esrc, const float* __restrict__ edst,
                 const _Float16* __restrict__ hhT, const float* __restrict__ tmax_p,
                 float* __restrict__ attn_out) {
    const int lane = threadIdx.x & 31;
    const int wave = threadIdx.x >> 5;
    const int task = blockIdx.x * 4 + wave;        // (N/16)*NH tasks
    const int head = task & (NH - 1);
    const int ib   = (task >> 2) << 4;
    const int r16  = lane & 15;
    const int row  = ib + r16;
    const bool hi  = lane >= 16;
    const int kb   = hi ? 8 : 0;                   // A-layout K base
    const int kb2  = hi ? 16 : 0;                  // unused; B loads use kb, same K order as A

    const float tmax = *tmax_p;
    const float esi  = esrc[head * N + row];
    const float* adj_row = adj  + (size_t)row * N;
    const float* tm_row  = tmat + (size_t)row * N;
    const float* ed      = edst + head * N;
    // B operand bases: d-column (r16 / r16+16) of h^T, K runs along n
    const _Float16* bT0 = hhT + ((size_t)head * HD + r16) * N;
    const _Float16* bT1 = hhT + ((size_t)head * HD + 16 + r16) * N;

    float m = -3.0e38f, s = 0.0f;
    v8f acc0 = {}; v8f acc1 = {};

    for (int jb = 0; jb < N; jb += 32) {
        // ---- vector loads: two aligned 32B runs per operand
        const v8f av_lo = *(const v8f*)(adj_row + jb + kb);
        const v8f av_hi = *(const v8f*)(adj_row + jb + 16 + kb);
        const v8f tm_lo = *(const v8f*)(tm_row  + jb + kb);
        const v8f tm_hi = *(const v8f*)(tm_row  + jb + 16 + kb);
        const v8f ed_lo = *(const v8f*)(ed      + jb + kb);
        const v8f ed_hi = *(const v8f*)(ed      + jb + 16 + kb);

        // ---- logits for this 16x32 tile, in A layout
        float lg[16];
        float tilemax = -3.0e38f;
#pragma unroll
        for (int t = 0; t < 16; ++t) {
            float av = (t < 8) ? av_lo[t] : av_hi[t - 8];
            float tv = (t < 8) ? tm_lo[t] : tm_hi[t - 8];
            float ev = (t < 8) ? ed_lo[t] : ed_hi[t - 8];
            float tw = __expf(-TD * (tmax - tv));
            float e  = esi + ev;
            e = (e > 0.0f ? e : ALPHA * e) * tw;
            float l = (av == 0.0f) ? -1.0e30f : e;
            lg[t] = l;
            tilemax = fmaxf(tilemax, l);
        }
        // row stats: row (lane&15) lives in this lane and lane^16
        tilemax = fmaxf(tilemax, __shfl_xor(tilemax, 16, 32));
        float mn   = fmaxf(m, tilemax);
        float psum = 0.0f;
        v16h pa;
#pragma unroll
        for (int t = 0; t < 16; ++t) {
            float p = __expf(lg[t] - mn);
            psum += p;
            pa[t] = (_Float16)p;
        }
        psum += __shfl_xor(psum, 16, 32);
        // ---- rescale only when some row's max actually moved (wave-uniform branch)
        if (__any(mn > m)) {
            float scale = __expf(m - mn);
            s *= scale;
#pragma unroll
            for (int r = 0; r < 8; ++r) {
                float sc = __shfl(scale, hi ? (r + 8) : r, 32);
                acc0[r] *= sc;
                acc1[r] *= sc;
            }
        }
        s += psum;
        m = mn;
        // ---- B operands: K slots {kb..kb+7, 16+kb..16+kb+7} per lane (two b128 loads each)
        typedef __attribute__((ext_vector_type(8))) _Float16 v8h_t;
        v16h b0 = __builtin_shufflevector(*(const v8h_t*)(bT0 + jb + kb), *(const v8h_t*)(bT0 + jb + 16 + kb), 0,1,2,3,4,5,6,7,8,9,10,11,12,13,14,15);
        v16h b1 = __builtin_shufflevector(*(const v8h_t*)(bT1 + jb + kb), *(const v8h_t*)(bT1 + jb + 16 + kb), 0,1,2,3,4,5,6,7,8,9,10,11,12,13,14,15);
        acc0 = __builtin_amdgcn_wmma_f32_16x16x32_f16(false, pa, false, b0, (short)0, acc0, false, false);
        acc1 = __builtin_amdgcn_wmma_f32_16x16x32_f16(false, pa, false, b1, (short)0, acc1, false, false);
    }
    // ---- normalize (single reciprocal per lane, shuffle the reciprocal) and write out
    float invs = 1.0f / s;
#pragma unroll
    for (int r = 0; r < 8; ++r) {
        int   orow = ib + (hi ? (r + 8) : r);
        float iv   = __shfl(invs, hi ? (r + 8) : r, 32);
        attn_out[(size_t)orow * HID + head * HD + r16]      = acc0[r] * iv;
        attn_out[(size_t)orow * HID + head * HD + 16 + r16] = acc1[r] * iv;
    }
}

// ---------------------------------------------------------------- ELU + LayerNorm
__global__ __launch_bounds__(HID)
void elu_ln_kernel(const float* __restrict__ attn, const float* __restrict__ g,
                   const float* __restrict__ b, float* __restrict__ xout) {
    int n = blockIdx.x, c = threadIdx.x;
    float v = attn[(size_t)n * HID + c];
    v = v > 0.0f ? v : (__expf(v) - 1.0f);
    __shared__ float red[HID];
    red[c] = v; __syncthreads();
    for (int s2 = HID / 2; s2 > 0; s2 >>= 1) { if (c < s2) red[c] += red[c + s2]; __syncthreads(); }
    float mu = red[0] / HID; __syncthreads();
    float dv = v - mu;
    red[c] = dv * dv; __syncthreads();
    for (int s2 = HID / 2; s2 > 0; s2 >>= 1) { if (c < s2) red[c] += red[c + s2]; __syncthreads(); }
    float var = red[0] / HID;
    xout[(size_t)n * HID + c] = dv * rsqrtf(var + EPS) * g[c] + b[c];
}

// ---------------------------------------------------------------- gather + MLP head
__global__ __launch_bounds__(HID)
void head_kernel(const float* __restrict__ x, const int* __restrict__ ids,
                 const float* __restrict__ attract, const float* __restrict__ Wa,
                 const float* __restrict__ ba, const float* __restrict__ W1,
                 const float* __restrict__ b1, const float* __restrict__ W2,
                 const float* __restrict__ b2, float* __restrict__ out) {
    int bb = blockIdx.x, k = threadIdx.x;
    __shared__ float tf[HID];
    int id   = ids[bb];
    float av = attract[bb];
    tf[k] = x[(size_t)id * HID + k] + av * Wa[k] + ba[k];
    __syncthreads();
    float hk = b1[k];
#pragma unroll 8
    for (int c = 0; c < HID; ++c) hk += tf[c] * W1[c * HID + k];
    hk = fmaxf(hk, 0.0f);
    __shared__ float red[HID];
    red[k] = hk * W2[k];
    __syncthreads();
    for (int s2 = HID / 2; s2 > 0; s2 >>= 1) { if (k < s2) red[k] += red[k + s2]; __syncthreads(); }
    if (k == 0) out[bb] = red[0] + b2[0];
}

// ---------------------------------------------------------------- launch
extern "C" void kernel_launch(void* const* d_in, const int* in_sizes, int n_in,
                              void* d_out, int out_size, void* d_ws, size_t ws_size,
                              hipStream_t stream) {
    const int*   topic_ids = (const int*)  d_in[0];
    const float* adj       = (const float*)d_in[1];
    const float* tmat      = (const float*)d_in[2];
    const float* attract   = (const float*)d_in[3];
    const float* emb       = (const float*)d_in[4];
    const float* Ws        = (const float*)d_in[5];
    const float* bs        = (const float*)d_in[6];
    const float* Wa        = (const float*)d_in[7];
    const float* ba        = (const float*)d_in[8];
    const float* gat_W     = (const float*)d_in[9];
    const float* gat_a     = (const float*)d_in[10];
    const float* ln_g      = (const float*)d_in[11];
    const float* ln_b      = (const float*)d_in[12];
    const float* W1        = (const float*)d_in[13];
    const float* b1        = (const float*)d_in[14];
    const float* W2        = (const float*)d_in[15];
    const float* b2        = (const float*)d_in[16];
    float* out = (float*)d_out;

    // workspace carve-up (256B aligned)
    char* p = (char*)d_ws;
    auto carve = [&](size_t bytes) { char* q = p; p += (bytes + 255) & ~(size_t)255; return q; };
    float*    tmax  = (float*)   carve(sizeof(float));
    float*    xA    = (float*)   carve((size_t)N * HID * 4);
    float*    xB    = (float*)   carve((size_t)N * HID * 4);
    float*    h     = (float*)   carve((size_t)NH * N * HD * 4);
    _Float16* hhT   = (_Float16*)carve((size_t)NH * N * HD * 2);
    float*    esrc  = (float*)   carve((size_t)NH * N * 4);
    float*    edst  = (float*)   carve((size_t)NH * N * 4);
    float*    attn  = (float*)   carve((size_t)N * HID * 4);

    // global max of time_matrix
    init_tmax_kernel<<<1, 32, 0, stream>>>(tmax);
    reduce_max_kernel<<<1024, 256, 0, stream>>>(tmat, N * N, tmax);

    // struct projection
    struct_kernel<<<(N * HID + 255) / 256, 256, 0, stream>>>(emb, Ws, bs, xA);

    float* xin = xA;
    float* xout = xB;
    for (int l = 0; l < L; ++l) {
        const float* Wl = gat_W + (size_t)l * NH * HID * HD;
        const float* al = gat_a + (size_t)l * NH * 2 * HD;
        proj_kernel<<<(NH * N * HD + 255) / 256, 256, 0, stream>>>(xin, Wl, h, hhT);
        edot_kernel<<<(NH * N + 255) / 256, 256, 0, stream>>>(h, al, esrc, edst);
        attn_kernel<<<(N / 16) * NH / 4, 128, 0, stream>>>(adj, tmat, esrc, edst, hhT, tmax, attn);
        elu_ln_kernel<<<N, HID, 0, stream>>>(attn, ln_g + l * HID, ln_b + l * HID, xout);
        float* t = xin; xin = xout; xout = t;
    }
    // xin now holds final node features
    head_kernel<<<B, HID, 0, stream>>>(xin, topic_ids, attract, Wa, ba, W1, b1, W2, b2, out);
}
